// Net_82145544503436
// MI455X (gfx1250) — 8-bit matrix operand order repaired and checked by reading, compiled, not run on hardware
//
#include <hip/hip_runtime.h>
#include <stdint.h>

typedef int v8i __attribute__((ext_vector_type(8)));
typedef int avi2 __attribute__((vector_size(8)));

#define BB 8
#define HH 256
#define WWI 256
#define CH 64
#define PITCH 72
static const size_t NPIX = (size_t)BB * HH * WWI;
static const size_t NACT = NPIX * CH;

#if defined(__gfx1250__) && __has_builtin(__builtin_amdgcn_global_load_async_to_lds_b64)
#define HAS_ASYNC_LDS 1
#else
#define HAS_ASYNC_LDS 0
#endif


__global__ __launch_bounds__(256) void binarize_input_kernel(
    const float* __restrict__ x, int8_t* __restrict__ sx) {
  int i = blockIdx.x * 256 + threadIdx.x;
  float v = x[i];
  sx[i] = (v > 0.f) ? 1 : ((v < 0.f) ? -1 : 0);
}

__global__ __launch_bounds__(256) void pack_w0_kernel(
    const float* __restrict__ w, int8_t* __restrict__ wp) {
  int i = blockIdx.x * 256 + threadIdx.x;
  if (i >= 64 * 9) return;
  float v = w[i];
  wp[i] = (v > 0.f) ? 1 : ((v < 0.f) ? -1 : 0);
}

__global__ __launch_bounds__(256) void pack_w4_kernel(
    const float* __restrict__ w, int8_t* __restrict__ wp) {
  int i = blockIdx.x * 256 + threadIdx.x;
  if (i >= 128 * 9) return;
  float v = w[i];
  wp[i] = (v > 0.f) ? 1 : ((v < 0.f) ? -1 : 0);
}

__global__ __launch_bounds__(256) void pack_w_kernel(
    const float* __restrict__ w, int8_t* __restrict__ wp) {
  int i = blockIdx.x * 256 + threadIdx.x;
  if (i >= 4 * 9 * 32 * 32) return;
  int byte = i & 31;
  int lane = (i >> 5) & 31;
  int idx  = i >> 10;
  int t  = idx % 9;
  int nt = idx / 9;
  int v4 = byte >> 2, j = byte & 3;
  int K = (v4 >> 1) * 16 + ((lane >= 16) ? 8 : 0) + (v4 & 1) * 4 + j;
  int N = lane & 15;
  int cout = nt * 16 + N;
  float val = w[(cout * 64 + K) * 9 + t];
  wp[i] = (val > 0.f) ? 1 : ((val < 0.f) ? -1 : 0);
}


__global__ __launch_bounds__(256) void conv0_kernel(
    const int8_t* __restrict__ sx, const int8_t* __restrict__ w0p,
    short* __restrict__ y) {
  int i = blockIdx.x * 256 + threadIdx.x;
  int w = i & 255, h = (i >> 8) & 255, b = i >> 16;
  int patch[9];
#pragma unroll
  for (int dy = 0; dy < 3; ++dy)
#pragma unroll
    for (int dx = 0; dx < 3; ++dx) {
      int hh = h - 1 + dy, ww = w - 1 + dx;
      patch[dy * 3 + dx] = (hh >= 0 && hh < HH && ww >= 0 && ww < WWI)
          ? (int)sx[(size_t)(b * HH + hh) * WWI + ww] : 0;
    }
  short* yo = y + (size_t)i * CH;
#pragma unroll 4
  for (int c = 0; c < CH; ++c) {
    int acc = 0;
#pragma unroll
    for (int t = 0; t < 9; ++t) acc += patch[t] * (int)w0p[c * 9 + t];
    yo[c] = (short)acc;
  }
}


__global__ __launch_bounds__(256) void conv_wmma_kernel(
    const int8_t* __restrict__ act,
    const int8_t* __restrict__ wpk,
    short* __restrict__ out,
    int* __restrict__ gsum, unsigned long long* __restrict__ gsq) {
  __shared__ __align__(16) int8_t lds[18 * 34 * PITCH];
  __shared__ int bsum[CH];
  __shared__ int bsq[CH];

  const int tid   = threadIdx.x;
  const int wg_w  = blockIdx.x * 32;
  const int hband = blockIdx.y * 16;
  const int b     = blockIdx.z;

  if (tid < CH) { bsum[tid] = 0; bsq[tid] = 0; }

  const bool interior = (blockIdx.x >= 1) && (blockIdx.x <= 6) &&
                        (blockIdx.y >= 1) && (blockIdx.y <= 14);
#if HAS_ASYNC_LDS
  if (interior) {
    for (int i = tid; i < 18 * 34 * 8; i += 256) {
      int r    = i / 272;
      int rem  = i - r * 272;
      int pix  = rem >> 3;
      int part = rem & 7;
      size_t goff = ((size_t)(b * HH + hband - 1 + r) * WWI + wg_w - 1 + pix) * CH
                    + (size_t)part * 8;
      int loff = (r * 34 + pix) * PITCH + part * 8;
      __builtin_amdgcn_global_load_async_to_lds_b64(
          (__attribute__((address_space(1))) avi2*)(act + goff),
          (__attribute__((address_space(3))) avi2*)(&lds[loff]), 0, 0);
    }
#if __has_builtin(__builtin_amdgcn_s_wait_asynccnt)
    __builtin_amdgcn_s_wait_asynccnt(0);
#else
    asm volatile("s_wait_asynccnt 0" ::: "memory");
#endif
  } else
#endif
  {
    for (int i = tid; i < 18 * 34 * 8; i += 256) {
      int r    = i / 272;
      int rem  = i - r * 272;
      int pix  = rem >> 3;
      int part = rem & 7;
      int hh = hband - 1 + r;
      int ww = wg_w - 1 + pix;
      int2 v = make_int2(0, 0);
      if (hh >= 0 && hh < HH && ww >= 0 && ww < WWI) {
        v = *reinterpret_cast<const int2*>(
            act + ((size_t)(b * HH + hh) * WWI + ww) * CH + part * 8);
      }
      *reinterpret_cast<int2*>(&lds[(r * 34 + pix) * PITCH + part * 8]) = v;
    }
  }

  const int wave  = tid >> 5;
  const int lane  = tid & 31;
  const int ctile = wave & 3;
  const int ptile = wave >> 2;
  const int mA    = lane & 15;
  const int koff  = (lane < 16) ? 0 : 8;

  v8i Bf[9];
#pragma unroll
  for (int t = 0; t < 9; ++t) {
    const int4* bp = reinterpret_cast<const int4*>(
        wpk + (((ctile * 9) + t) * 32 + lane) * 32);
    int4 lo = bp[0], hi = bp[1];
    Bf[t][0] = lo.x; Bf[t][1] = lo.y; Bf[t][2] = lo.z; Bf[t][3] = lo.w;
    Bf[t][4] = hi.x; Bf[t][5] = hi.y; Bf[t][6] = hi.z; Bf[t][7] = hi.w;
  }

  __syncthreads();

  const int n    = lane & 15;
  const int c    = ctile * 16 + n;
  const int mtop = (lane >> 4) << 3;

  int ssum = 0, ssq = 0;

#pragma unroll 2
  for (int hr = 0; hr < 16; ++hr) {
    v8i acc = {0, 0, 0, 0, 0, 0, 0, 0};
#pragma unroll
    for (int dy = 0; dy < 3; ++dy) {
#pragma unroll
      for (int dx = 0; dx < 3; ++dx) {
        const int pix = ptile * 16 + mA + dx;
        const int8_t* ab = &lds[((hr + dy) * 34 + pix) * PITCH + koff];
        int2 c0 = *reinterpret_cast<const int2*>(ab + 0);
        int2 c1 = *reinterpret_cast<const int2*>(ab + 16);
        int2 c2 = *reinterpret_cast<const int2*>(ab + 32);
        int2 c3 = *reinterpret_cast<const int2*>(ab + 48);
        v8i A;
        A[0] = c0.x; A[1] = c0.y; A[2] = c1.x; A[3] = c1.y;
        A[4] = c2.x; A[5] = c2.y; A[6] = c3.x; A[7] = c3.y;
        acc = __builtin_amdgcn_wmma_i32_16x16x64_iu8(
            true, A, true, Bf[dy * 3 + dx], acc, false, false);
      }
    }
    size_t obase = ((size_t)(b * HH + hband + hr) * WWI + wg_w + ptile * 16 + mtop)
                   * CH + c;
#pragma unroll
    for (int r = 0; r < 8; ++r) {
      int v = acc[r];
      out[obase + (size_t)r * CH] = (short)v;
      ssum += v;
      ssq  += v * v;
    }
  }

  atomicAdd(&bsum[c], ssum);
  atomicAdd(&bsq[c], ssq);
  __syncthreads();
  if (tid < CH) {
    atomicAdd(&gsum[tid], bsum[tid]);
    atomicAdd(&gsq[tid], (unsigned long long)(unsigned int)bsq[tid]);
  }
}


__global__ __launch_bounds__(256) void stats_kernel(
    const short* __restrict__ y, int* __restrict__ gsum,
    unsigned long long* __restrict__ gsq) {
  __shared__ int ssum[CH];
  __shared__ unsigned long long ssq[CH];
  int tid = threadIdx.x;
  if (tid < CH) { ssum[tid] = 0; ssq[tid] = 0ull; }
  __syncthreads();
  size_t base = (size_t)blockIdx.x * 16384 + tid;
  int s = 0;
  unsigned long long q = 0ull;
  for (int j = 0; j < 64; ++j) {
    int v = (int)y[base + (size_t)j * 256];
    s += v;
    q += (unsigned long long)(v * v);
  }
  atomicAdd(&ssum[tid & 63], s);
  atomicAdd(&ssq[tid & 63], q);
  __syncthreads();
  if (tid < CH) {
    atomicAdd(&gsum[tid], ssum[tid]);
    atomicAdd(&gsq[tid], ssq[tid]);
  }
}

__global__ void thresh_kernel(
    const int* __restrict__ gsum, const unsigned long long* __restrict__ gsq,
    const float* __restrict__ g, const float* __restrict__ be,
    float* __restrict__ T, float* __restrict__ flip) {
  int c = threadIdx.x;
  if (c >= CH) return;
  const double N = 524288.0;
  double mc  = (double)gsum[c] / N;
  double var = (double)gsq[c] / N - mc * mc;
  double s   = sqrt(var + 1e-5);
  float gc = g[c], bec = be[c];
  float t, fl;
  if (gc != 0.f) {
    t  = (float)(mc - (double)bec * s / (double)gc);
    fl = (gc > 0.f) ? 1.f : -1.f;
  } else {
    t  = -1e30f;
    fl = (bec > 0.f) ? 1.f : ((bec < 0.f) ? -1.f : 0.f);
  }
  T[c] = t; flip[c] = fl;
}

__global__ __launch_bounds__(256) void binarize_kernel(
    const short* __restrict__ y, const float* __restrict__ T,
    const float* __restrict__ flip, int8_t* __restrict__ a) {
  size_t i0 = ((size_t)blockIdx.x * 256 + threadIdx.x) * 4;
#pragma unroll
  for (int k = 0; k < 4; ++k) {
    size_t i = i0 + k;
    int c = (int)(i & 63);
    float fl = flip[c];
    float r = fl * (((float)y[i] > T[c]) ? 1.f : -1.f);
    a[i] = (int8_t)r;
  }
}


__global__ __launch_bounds__(256) void conv4_kernel(
    const int8_t* __restrict__ a1, const int8_t* __restrict__ a4,
    const int8_t* __restrict__ w4p, const float* __restrict__ x,
    const float* __restrict__ b4, float* __restrict__ out) {
  int i = blockIdx.x * 256 + threadIdx.x;
  int w = i & 255, h = (i >> 8) & 255, b = i >> 16;
  int acc = 0;
#pragma unroll
  for (int dy = 0; dy < 3; ++dy) {
    int hh = h - 1 + dy;
    if (hh < 0 || hh >= HH) continue;
#pragma unroll
    for (int dx = 0; dx < 3; ++dx) {
      int ww = w - 1 + dx;
      if (ww < 0 || ww >= WWI) continue;
      size_t pb = ((size_t)(b * HH + hh) * WWI + ww) * CH;
      const int8_t* p1 = a1 + pb;
      const int8_t* p4 = a4 + pb;
      int t = dy * 3 + dx;
#pragma unroll 8
      for (int c = 0; c < CH; ++c) acc += (int)p1[c] * (int)w4p[c * 9 + t];
#pragma unroll 8
      for (int c = 0; c < CH; ++c) acc += (int)p4[c] * (int)w4p[(CH + c) * 9 + t];
    }
  }
  out[i] = x[i] + (float)acc + b4[0];
}


extern "C" void kernel_launch(void* const* d_in, const int* in_sizes, int n_in,
                              void* d_out, int out_size, void* d_ws, size_t ws_size,
                              hipStream_t stream) {
  const float* x   = (const float*)d_in[0];
  const float* w0  = (const float*)d_in[1];
  const float* g0  = (const float*)d_in[3];
  const float* be0 = (const float*)d_in[4];
  const float* w1  = (const float*)d_in[5];
  const float* g1  = (const float*)d_in[7];
  const float* be1 = (const float*)d_in[8];
  const float* w2  = (const float*)d_in[9];
  const float* g2  = (const float*)d_in[11];
  const float* be2 = (const float*)d_in[12];
  const float* w3  = (const float*)d_in[13];
  const float* g3  = (const float*)d_in[15];
  const float* be3 = (const float*)d_in[16];
  const float* w4  = (const float*)d_in[17];
  const float* b4  = (const float*)d_in[18];
  float* out = (float*)d_out;

  uint8_t* ws = (uint8_t*)d_ws;
  size_t off = 0;
  auto walloc = [&](size_t sz) -> void* {
    void* p = ws + off;
    off += (sz + 255) & ~(size_t)255;
    return p;
  };
  int8_t* sx  = (int8_t*)walloc(NPIX);
  int8_t* w0p = (int8_t*)walloc(64 * 9);
  int8_t* w4p = (int8_t*)walloc(128 * 9);
  int8_t* wpk = (int8_t*)walloc(4 * 9 * 32 * 32);
  int*    gsm = (int*)walloc(CH * sizeof(int));
  unsigned long long* gsq = (unsigned long long*)walloc(CH * 8);
  float* thrT = (float*)walloc(CH * 4);
  float* thrF = (float*)walloc(CH * 4);
  short*  y16 = (short*)walloc(NACT * 2);
  int8_t* a1  = (int8_t*)walloc(NACT);
  int8_t* aA  = (int8_t*)walloc(NACT);
  int8_t* aB  = (int8_t*)walloc(NACT);

  binarize_input_kernel<<<(int)(NPIX / 256), 256, 0, stream>>>(x, sx);

  pack_w0_kernel<<<3, 256, 0, stream>>>(w0, w0p);
  conv0_kernel<<<(int)(NPIX / 256), 256, 0, stream>>>(sx, w0p, y16);
  (void)hipMemsetAsync(gsm, 0, CH * sizeof(int), stream);
  (void)hipMemsetAsync(gsq, 0, CH * 8, stream);
  stats_kernel<<<(int)(NACT / 16384), 256, 0, stream>>>(y16, gsm, gsq);
  thresh_kernel<<<1, 64, 0, stream>>>(gsm, gsq, g0, be0, thrT, thrF);
  binarize_kernel<<<(int)(NACT / 1024), 256, 0, stream>>>(y16, thrT, thrF, a1);

  auto layerW = [&](const float* wN, const float* g, const float* be,
                    const int8_t* ain, int8_t* aout) {
    pack_w_kernel<<<144, 256, 0, stream>>>(wN, wpk);
    (void)hipMemsetAsync(gsm, 0, CH * sizeof(int), stream);
    (void)hipMemsetAsync(gsq, 0, CH * 8, stream);
    conv_wmma_kernel<<<dim3(8, 16, 8), 256, 0, stream>>>(ain, wpk, y16, gsm, gsq);
    thresh_kernel<<<1, 64, 0, stream>>>(gsm, gsq, g, be, thrT, thrF);
    binarize_kernel<<<(int)(NACT / 1024), 256, 0, stream>>>(y16, thrT, thrF, aout);
  };
  layerW(w1, g1, be1, a1, aA);
  layerW(w2, g2, be2, aA, aB);
  layerW(w3, g3, be3, aB, aA);

  pack_w4_kernel<<<5, 256, 0, stream>>>(w4, w4p);
  conv4_kernel<<<(int)(NPIX / 256), 256, 0, stream>>>(a1, aA, w4p, x, b4, out);
}
